// GCNII_21827023798524
// MI455X (gfx1250) — compile-verified, matrix operand order checked by reading, not run on hardware
//
#include <hip/hip_runtime.h>
#include <hip/hip_bf16.h>
#include <cmath>

#define NN      10000
#define EE      320000
#define NFEAT_  512
#define NHID_   256
#define NCLASS_ 64
#define NLAYERS 8
#define ALPHA_  0.1f
#define BN_EPS_ 1e-5f
#define NBLK    79   // ceil(10000/128)

typedef __attribute__((ext_vector_type(16))) __bf16 bf16x16;
typedef __attribute__((ext_vector_type(8)))  __bf16 bf16x8;
typedef __attribute__((ext_vector_type(8)))  float  v8f;

// ---------------- elementwise / setup kernels ----------------

__global__ __launch_bounds__(256) void cast_bf16_k(const float* __restrict__ s,
                                                   __bf16* __restrict__ d, int n) {
    int t = blockIdx.x * blockDim.x + threadIdx.x;
    if (t < n) d[t] = (__bf16)s[t];
}

// Pack row-major f32 weight [K x Nn] into WMMA-B lane layout (bf16):
// packed[((kt*tilesN + nt)*32 + lane)*16 + e] = B[k][n]
// with h = lane>>4, kk = (e<8 ? h*8+e : 16+h*8+(e-8)), k = kt*32+kk, n = nt*16+(lane&15)
__global__ __launch_bounds__(256) void pack_b_k(const float* __restrict__ src,
                                                __bf16* __restrict__ dst, int K, int Nn) {
    int tid = blockIdx.x * blockDim.x + threadIdx.x;
    int total = K * Nn;
    if (tid >= total) return;
    int e    = tid & 15;
    int lane = (tid >> 4) & 31;
    int rest = tid >> 9;
    int tilesN = Nn >> 4;
    int nt = rest % tilesN;
    int kt = rest / tilesN;
    int h  = lane >> 4;
    int kk = (e < 8) ? (h * 8 + e) : (16 + h * 8 + (e - 8));
    int k  = kt * 32 + kk;
    int n  = nt * 16 + (lane & 15);
    dst[tid] = (__bf16)src[(size_t)k * Nn + n];
}

__global__ __launch_bounds__(256) void deg_k(const int* __restrict__ col,
                                             float* __restrict__ deg) {
    int e = blockIdx.x * blockDim.x + threadIdx.x;
    if (e < EE) atomicAdd(&deg[col[e]], 1.0f);
}

__global__ __launch_bounds__(256) void dinv_k(const float* __restrict__ deg,
                                              float* __restrict__ dinv) {
    int i = blockIdx.x * blockDim.x + threadIdx.x;
    if (i < NN) dinv[i] = rsqrtf(deg[i] + 1.0f);   // +1 for self loop; always > 0
}

// ---------------- SpMM: p[col] += dinv[row]*dinv[col]*h[row] ----------------
// 64 threads per edge, float4 per thread: b128 gather + 4x f32 L2 atomics
__global__ __launch_bounds__(256) void spmm_edges_k(const int* __restrict__ rowi,
                                                    const int* __restrict__ coli,
                                                    const float* __restrict__ dinv,
                                                    const float* __restrict__ h,
                                                    float* __restrict__ p) {
    long tid = (long)blockIdx.x * 256 + threadIdx.x;
    int e = (int)(tid >> 6);
    int q = (int)(tid & 63);
    if (e >= EE) return;
    int r = rowi[e], c = coli[e];
    float w = dinv[r] * dinv[c];
    const float4 hv = *(const float4*)(h + (size_t)r * NHID_ + q * 4);
    float* dst = p + (size_t)c * NHID_ + q * 4;
    atomicAdd(dst + 0, w * hv.x);
    atomicAdd(dst + 1, w * hv.y);
    atomicAdd(dst + 2, w * hv.z);
    atomicAdd(dst + 3, w * hv.w);
}

// p = (1-alpha)*(p_edges + dinv^2 * h) + alpha*x0 ; also emit bf16 copy for WMMA A
__global__ __launch_bounds__(256) void blend_k(float* __restrict__ p,
                                               const float* __restrict__ h,
                                               const float* __restrict__ x0,
                                               const float* __restrict__ dinv,
                                               __bf16* __restrict__ pb) {
    int tid = blockIdx.x * blockDim.x + threadIdx.x;
    if (tid >= NN * NHID_) return;
    int m = tid >> 8;  // NHID_ == 256
    float di = dinv[m];
    float v = (1.0f - ALPHA_) * (p[tid] + di * di * h[tid]) + ALPHA_ * x0[tid];
    p[tid]  = v;
    pb[tid] = (__bf16)v;
}

// ---------------- WMMA GEMM: one wave = 16x64 tile (4 N-tiles register-blocked) ----
// A fragment loaded once per K-step, 4 WMMAs issued against it.
// mode 0: out0 = relu(acc + bias), out1 = out0   (lin0, builds x0)
// mode 1: out0 = (1-beta)*Pres + beta*acc        (GCN2 identity mapping)
// mode 2: out0 = acc + bias                      (lin1 -> logits)
__global__ __launch_bounds__(256)
void gemm_wmma_k(const __bf16* __restrict__ A, const __bf16* __restrict__ Bp,
                 int M, int K, int Nn, int mode,
                 const float* __restrict__ bias, const float* __restrict__ Pres,
                 float betaCoef, float* __restrict__ out0, float* __restrict__ out1) {
    const int lane  = threadIdx.x & 31;
    const int wave  = threadIdx.x >> 5;
    const int tilesN = Nn >> 4;
    const int quads  = tilesN >> 2;             // groups of 4 N-tiles (64 cols)
    const int job = blockIdx.x * 8 + wave;
    if (job >= (M >> 4) * quads) return;        // whole wave exits -> EXEC all-1s at WMMA
    const int tm = job / quads;
    const int tq = job % quads;
    const int half = lane >> 4;
    const int mrow = tm * 16 + (lane & 15);

    const __bf16* Arow  = A + (size_t)mrow * K + half * 8;
    // packed B: ((kt*tilesN + nt)*32 + lane)*16 ; nt = tq*4 + q
    const __bf16* Bbase = Bp + ((size_t)(tq * 4) * 32 + lane) * 16;
    const size_t bstride = (size_t)tilesN * 512;   // elements per K-tile
    const int ktiles = K >> 5;

    v8f acc0 = {0.f,0.f,0.f,0.f,0.f,0.f,0.f,0.f};
    v8f acc1 = acc0, acc2 = acc0, acc3 = acc0;

    for (int kt = 0; kt < ktiles; ++kt) {
        bf16x8 a0 = *(const bf16x8*)(Arow + (size_t)kt * 32);        // K = h*8 .. h*8+7
        bf16x8 a1 = *(const bf16x8*)(Arow + (size_t)kt * 32 + 16);   // K = 16+h*8 ..
        bf16x16 a = __builtin_shufflevector(a0, a1,
                       0, 1, 2, 3, 4, 5, 6, 7, 8, 9, 10, 11, 12, 13, 14, 15);
        const __bf16* bkt = Bbase + (size_t)kt * bstride;
        bf16x16 b0 = *(const bf16x16*)(bkt);
        bf16x16 b1 = *(const bf16x16*)(bkt + 512);
        bf16x16 b2 = *(const bf16x16*)(bkt + 1024);
        bf16x16 b3 = *(const bf16x16*)(bkt + 1536);
        acc0 = __builtin_amdgcn_wmma_f32_16x16x32_bf16(false, a, false, b0, (short)0, acc0, false, false);
        acc1 = __builtin_amdgcn_wmma_f32_16x16x32_bf16(false, a, false, b1, (short)0, acc1, false, false);
        acc2 = __builtin_amdgcn_wmma_f32_16x16x32_bf16(false, a, false, b2, (short)0, acc2, false, false);
        acc3 = __builtin_amdgcn_wmma_f32_16x16x32_bf16(false, a, false, b3, (short)0, acc3, false, false);
    }

    const int nbase = tq * 64 + (lane & 15);
#pragma unroll
    for (int q = 0; q < 4; ++q) {
        const v8f acc = (q == 0) ? acc0 : (q == 1) ? acc1 : (q == 2) ? acc2 : acc3;
        const int ncol = nbase + q * 16;
#pragma unroll
        for (int r = 0; r < 8; ++r) {
            int mg = tm * 16 + half * 8 + r;   // C/D layout: lanes 16-31 hold M=8..15
            size_t idx = (size_t)mg * Nn + ncol;
            float v = acc[r];
            if (mode == 0) {
                v += bias[ncol];
                v = v > 0.f ? v : 0.f;
                out0[idx] = v;
                out1[idx] = v;
            } else if (mode == 1) {
                out0[idx] = (1.0f - betaCoef) * Pres[idx] + betaCoef * v;
            } else {
                out0[idx] = v + bias[ncol];
            }
        }
    }
}

// ---------------- BatchNorm over nodes (axis 0) ----------------

__global__ __launch_bounds__(256) void bn_partial_k(const float* __restrict__ o,
                                                    float* __restrict__ ps,
                                                    float* __restrict__ ps2) {
    int f  = threadIdx.x;
    int m0 = blockIdx.x * 128;
    int m1 = m0 + 128 < NN ? m0 + 128 : NN;
    float s = 0.f, s2 = 0.f;
    for (int m = m0; m < m1; ++m) {
        float v = o[(size_t)m * NHID_ + f];
        s += v; s2 += v * v;
    }
    ps [blockIdx.x * NHID_ + f] = s;
    ps2[blockIdx.x * NHID_ + f] = s2;
}

__global__ __launch_bounds__(256) void bn_final_k(const float* __restrict__ ps,
                                                  const float* __restrict__ ps2,
                                                  float* __restrict__ mu,
                                                  float* __restrict__ rstd) {
    int f = threadIdx.x;
    float s = 0.f, s2 = 0.f;
    for (int b = 0; b < NBLK; ++b) {
        s  += ps [b * NHID_ + f];
        s2 += ps2[b * NHID_ + f];
    }
    float mean = s / (float)NN;
    float var  = s2 / (float)NN - mean * mean;
    mu[f]   = mean;
    rstd[f] = rsqrtf(var + BN_EPS_);
}

__global__ __launch_bounds__(256) void bn_norm_relu_k(const float* __restrict__ o,
                                                      const float* __restrict__ mu,
                                                      const float* __restrict__ rstd,
                                                      const float* __restrict__ g,
                                                      const float* __restrict__ bb,
                                                      float* __restrict__ h) {
    int tid = blockIdx.x * blockDim.x + threadIdx.x;
    if (tid >= NN * NHID_) return;
    int f = tid & (NHID_ - 1);
    float v = (o[tid] - mu[f]) * rstd[f] * g[f] + bb[f];
    h[tid] = v > 0.f ? v : 0.f;
}

// ---------------- host ----------------

static inline void* carve(char*& base, size_t bytes) {
    void* p = base;
    base += (bytes + 255) & ~(size_t)255;
    return p;
}

extern "C" void kernel_launch(void* const* d_in, const int* in_sizes, int n_in,
                              void* d_out, int out_size, void* d_ws, size_t ws_size,
                              hipStream_t stream) {
    (void)in_sizes; (void)n_in; (void)out_size; (void)ws_size;
    const float* x    = (const float*)d_in[0];
    const int*   ei   = (const int*)  d_in[1];
    const float* w0   = (const float*)d_in[2];
    const float* b0   = (const float*)d_in[3];
    const float* w1   = (const float*)d_in[4];
    const float* b1   = (const float*)d_in[5];
    const float* wc   = (const float*)d_in[6];
    const float* gam  = (const float*)d_in[7];
    const float* bet  = (const float*)d_in[8];
    float*       out  = (float*)d_out;
    const int* erow = ei;
    const int* ecol = ei + EE;

    char* wsp = (char*)d_ws;
    float*  deg   = (float*) carve(wsp, (size_t)NN * 4);
    float*  dinv  = (float*) carve(wsp, (size_t)NN * 4);
    __bf16* xb    = (__bf16*)carve(wsp, (size_t)NN * NFEAT_ * 2);
    __bf16* w0p   = (__bf16*)carve(wsp, (size_t)NFEAT_ * NHID_ * 2);
    __bf16* wcp   = (__bf16*)carve(wsp, (size_t)NLAYERS * NHID_ * NHID_ * 2);
    __bf16* w1p   = (__bf16*)carve(wsp, (size_t)NHID_ * NCLASS_ * 2);
    float*  x0    = (float*) carve(wsp, (size_t)NN * NHID_ * 4);
    float*  hbuf  = (float*) carve(wsp, (size_t)NN * NHID_ * 4);
    float*  pbuf  = (float*) carve(wsp, (size_t)NN * NHID_ * 4);
    __bf16* pb16  = (__bf16*)carve(wsp, (size_t)NN * NHID_ * 2);
    float*  obuf  = (float*) carve(wsp, (size_t)NN * NHID_ * 4);
    float*  mu    = (float*) carve(wsp, (size_t)NHID_ * 4);
    float*  rstd  = (float*) carve(wsp, (size_t)NHID_ * 4);
    float*  psum  = (float*) carve(wsp, (size_t)NBLK * NHID_ * 4);
    float*  psum2 = (float*) carve(wsp, (size_t)NBLK * NHID_ * 4);

    // degrees -> dinv
    hipMemsetAsync(deg, 0, (size_t)NN * 4, stream);
    deg_k<<<(EE + 255) / 256, 256, 0, stream>>>(ecol, deg);
    dinv_k<<<(NN + 255) / 256, 256, 0, stream>>>(deg, dinv);

    // one-time casts / weight pre-swizzle
    cast_bf16_k<<<((size_t)NN * NFEAT_ + 255) / 256, 256, 0, stream>>>(x, xb, NN * NFEAT_);
    pack_b_k<<<(NFEAT_ * NHID_ + 255) / 256, 256, 0, stream>>>(w0, w0p, NFEAT_, NHID_);
    for (int l = 0; l < NLAYERS; ++l)
        pack_b_k<<<(NHID_ * NHID_ + 255) / 256, 256, 0, stream>>>(
            wc + (size_t)l * NHID_ * NHID_, wcp + (size_t)l * NHID_ * NHID_, NHID_, NHID_);
    pack_b_k<<<(NHID_ * NCLASS_ + 255) / 256, 256, 0, stream>>>(w1, w1p, NHID_, NCLASS_);

    // h = relu(x @ W0 + b0); x0 = h        (jobs = 625 * (256/64) = 2500)
    const int jobsH = (NN / 16) * ((NHID_ / 16) / 4);
    gemm_wmma_k<<<(jobsH + 7) / 8, 256, 0, stream>>>(
        xb, w0p, NN, NFEAT_, NHID_, 0, b0, nullptr, 0.f, hbuf, x0);

    const long spmmThreads = (long)EE * 64;
    for (int l = 0; l < NLAYERS; ++l) {
        float betaC = logf(0.5f / (float)(l + 1) + 1.0f);
        hipMemsetAsync(pbuf, 0, (size_t)NN * NHID_ * 4, stream);
        spmm_edges_k<<<(int)((spmmThreads + 255) / 256), 256, 0, stream>>>(
            erow, ecol, dinv, hbuf, pbuf);
        blend_k<<<((size_t)NN * NHID_ + 255) / 256, 256, 0, stream>>>(
            pbuf, hbuf, x0, dinv, pb16);
        gemm_wmma_k<<<(jobsH + 7) / 8, 256, 0, stream>>>(
            pb16, wcp + (size_t)l * NHID_ * NHID_, NN, NHID_, NHID_, 1,
            nullptr, pbuf, betaC, obuf, nullptr);
        bn_partial_k<<<NBLK, 256, 0, stream>>>(obuf, psum, psum2);
        bn_final_k<<<1, 256, 0, stream>>>(psum, psum2, mu, rstd);
        bn_norm_relu_k<<<((size_t)NN * NHID_ + 255) / 256, 256, 0, stream>>>(
            obuf, mu, rstd, gam + (size_t)l * NHID_, bet + (size_t)l * NHID_, hbuf);
    }

    // logits = h @ W1 + b1                  (jobs = 625 * (64/16)/4 = 625)
    cast_bf16_k<<<((size_t)NN * NHID_ + 255) / 256, 256, 0, stream>>>(hbuf, pb16, NN * NHID_);
    const int jobsC = (NN / 16) * ((NCLASS_ / 16) / 4);
    gemm_wmma_k<<<(jobsC + 7) / 8, 256, 0, stream>>>(
        pb16, w1p, NN, NHID_, NCLASS_, 2, b1, nullptr, 0.f, out, nullptr);
}
